// PMGTSelfAttention_79568564126321
// MI455X (gfx1250) — matrix operand order repaired and checked by reading, not compiled, not run on hardware
//
#include <hip/hip_runtime.h>
#include <hip/hip_bf16.h>

// ---------------------------------------------------------------------------
// PMGT self-attention with relative_key_query bias, bf16 WMMA on gfx1250.
//   B=4, S=1024, HIDDEN=1024, 16 heads x 64, MAX_POS=1024.
// Kernel 1: QKV projection GEMM (f32 in -> bf16 out), WMMA 16x16x32 bf16.
// Kernel 2: flash-style attention with WMMA score / bias-band / PV matmuls;
//           K / V^T tiles staged into LDS via the Tensor Data Mover when the
//           toolchain exposes the tensor_load_to_lds builtin (5- or 6-arg form).
// ---------------------------------------------------------------------------

typedef __attribute__((ext_vector_type(16))) __bf16 bfx16;
typedef __attribute__((ext_vector_type(8)))  float  fx8;
typedef unsigned int u32x4 __attribute__((ext_vector_type(4)));
typedef int          i32x8 __attribute__((ext_vector_type(8)));
typedef int          i32x4 __attribute__((ext_vector_type(4)));

#define HS   1024
#define NH   16
#define HD   64
#define SEQ  1024

#if defined(__AMDGCN__) && __has_builtin(__builtin_amdgcn_tensor_load_to_lds) && \
    __has_builtin(__builtin_amdgcn_s_wait_tensorcnt)
  #if (__clang_major__ <= 22)
    #define TDM_MODE 5
  #else
    #define TDM_MODE 6
  #endif
#else
  #define TDM_MODE 0
#endif

union FragU { bfx16 v; uint4 q[2]; };

__device__ __forceinline__ fx8 wmma_bf16(bfx16 a, bfx16 b, fx8 c) {
  return __builtin_amdgcn_wmma_f32_16x16x32_bf16(false, a, false, b,
                                                 (short)0, c, false, false);
}

// A fragment (16x32 bf16, MxK): lane%16 = row M, lane/16 picks K half.
// lane holds K = {klo..klo+7, klo+16..klo+23}, klo = (lane/16)*8.
__device__ __forceinline__ bfx16 load_frag_a(const __bf16* p, int ld) {
  const int lane = threadIdx.x & 31;
  const int row  = lane & 15;
  const int klo  = (lane >> 4) * 8;
  FragU f;
  f.q[0] = *(const uint4*)(p + row * ld + klo);
  f.q[1] = *(const uint4*)(p + row * ld + klo + 16);
  return f.v;
}

// B fragment (32x16 bf16, KxN) loaded from B^T stored row-major [N][K]:
// lane%16 = col N, lane holds K = {k0..k0+7, k0+16..k0+23}, k0 = (lane/16)*8.
__device__ __forceinline__ bfx16 load_frag_b(const __bf16* p, int ld) {
  const int lane = threadIdx.x & 31;
  const int col  = lane & 15;
  const int k0   = (lane >> 4) * 8;
  FragU f;
  f.q[0] = *(const uint4*)(p + col * ld + k0);
  f.q[1] = *(const uint4*)(p + col * ld + k0 + 16);
  return f.v;
}

// C/D fragment (16x16 f32): lane%16 = col N, rows M = (lane/16)*8 + j.
__device__ __forceinline__ void store_frag_c_bf16(__bf16* p, int ld, fx8 c) {
  const int lane = threadIdx.x & 31;
  const int n  = lane & 15;
  const int mb = (lane >> 4) * 8;
#pragma unroll
  for (int j = 0; j < 8; ++j) p[(mb + j) * ld + n] = (__bf16)c[j];
}

__device__ __forceinline__ void cvt8(float4 a, float4 b, __bf16* d) {
  d[0]=(__bf16)a.x; d[1]=(__bf16)a.y; d[2]=(__bf16)a.z; d[3]=(__bf16)a.w;
  d[4]=(__bf16)b.x; d[5]=(__bf16)b.y; d[6]=(__bf16)b.z; d[7]=(__bf16)b.w;
}

__device__ __forceinline__ fx8 zero8() { fx8 z = {0.f,0.f,0.f,0.f,0.f,0.f,0.f,0.f}; return z; }

#if TDM_MODE
// ---------------------------------------------------------------------------
// TDM: DMA a 2D tile (bf16) global -> LDS, with LDS padding to reproduce the
// software pitch.  D# per CDNA5 ISA:
//   group0: count=1 | lds_addr | global_addr(57b) | type=2
//   group1: data_size=1(2B), pad_enable, pad_interval/amount codes,
//           tensor_dim0/1, tile_dim0/1, tensor_dim0_stride (48b)
// pad_interval code: 0=2,1=4,2=8,3=16,4=32... DWORDs; pad_amount code: n -> n+1 DWORDs.
// ---------------------------------------------------------------------------
__device__ __forceinline__ void tdm_load_tile_2d(
    unsigned lds_off, const void* gptr,
    unsigned tensor_d0, unsigned tensor_d1,
    unsigned tile_d0,   unsigned tile_d1,
    unsigned long long stride0_elems,
    unsigned pad_interval_code, unsigned pad_amount_code)
{
  const unsigned long long ga = (unsigned long long)(uintptr_t)gptr;
  u32x4 g0;
  g0[0] = 1u;                                            // count=1 (valid user D#)
  g0[1] = lds_off;                                       // LDS byte address
  g0[2] = (unsigned)ga;                                  // global_addr[31:0]
  g0[3] = (unsigned)((ga >> 32) & 0x01FFFFFFu) | (2u << 30);  // addr[56:32] | type=2
  i32x8 g1;
  g1[0] = (int)((1u << 16) |                             // data_size = 2 bytes
                (1u << 20) |                             // pad_enable
                (pad_interval_code << 22) |
                (pad_amount_code   << 25));
  g1[1] = (int)((tensor_d0 & 0xFFFFu) << 16);            // tensor_dim0[15:0] @ [63:48]
  g1[2] = (int)(((tensor_d0 >> 16) & 0xFFFFu) |          // tensor_dim0[31:16]
                ((tensor_d1 & 0xFFFFu) << 16));          // tensor_dim1[15:0]
  g1[3] = (int)(((tensor_d1 >> 16) & 0xFFFFu) |          // tensor_dim1[31:16]
                ((tile_d0 & 0xFFFFu) << 16));            // tile_dim0
  g1[4] = (int)(tile_d1 & 0xFFFFu);                      // tile_dim1 (tile_dim2=0)
  g1[5] = (int)(unsigned)(stride0_elems & 0xFFFFFFFFull);// stride0[31:0]
  g1[6] = (int)(unsigned)((stride0_elems >> 32) & 0xFFFFull); // stride0[47:32], stride1=0
  g1[7] = 0;
  const i32x4 gz = {0, 0, 0, 0};                         // 2D tensor: groups 2/3 unused
#if TDM_MODE == 5
  __builtin_amdgcn_tensor_load_to_lds(g0, g1, gz, gz, 0);
#else
  const i32x8 gz8 = {0, 0, 0, 0, 0, 0, 0, 0};
  __builtin_amdgcn_tensor_load_to_lds(g0, g1, gz, gz, gz8, 0);
#endif
}
#endif  // TDM_MODE

// ---------------------------------------------------------------------------
// Kernel 1: QKV projection.  out[t,o] = sum_c hs[t,c] * W[o,c] + bias[o]
// grid.x: token tiles of 128 (M=4096), grid.y: 48 = {q,k,v} x 16 heads (N=64).
// Q,K stored [B,nH,S,64] bf16 (row d-contiguous). V stored [B,nH,64,S] bf16.
// ---------------------------------------------------------------------------
__global__ __launch_bounds__(256) void qkv_proj_kernel(
    const float* __restrict__ hs,
    const float* __restrict__ Wq, const float* __restrict__ Wk,
    const float* __restrict__ Wv,
    const float* __restrict__ bq, const float* __restrict__ bk,
    const float* __restrict__ bv,
    __bf16* __restrict__ Qh, __bf16* __restrict__ Kh, __bf16* __restrict__ Vt)
{
  const int t0   = blockIdx.x * 128;
  const int y    = blockIdx.y;        // 0..47
  const int mat  = y >> 4;            // 0=Q 1=K 2=V
  const int h    = y & 15;
  const int o0   = h * HD;            // output column base within 1024

  const float* W    = (mat == 0) ? Wq : (mat == 1) ? Wk : Wv;
  const float* bias = (mat == 0) ? bq : (mat == 1) ? bk : bv;

  __shared__ __bf16 As[128][40];      // 128x32 tile, padded pitch
  __shared__ __bf16 Bs[64][40];       // 64x32 tile

  const int tid  = threadIdx.x;
  const int lane = tid & 31;
  const int wave = tid >> 5;          // 8 waves
  const int mw   = wave >> 1;         // 0..3 : 32-row strip
  const int nw   = wave & 1;          // 0..1 : 32-col strip

  fx8 acc[2][2];
#pragma unroll
  for (int i = 0; i < 2; ++i)
#pragma unroll
    for (int j = 0; j < 2; ++j) acc[i][j] = zero8();

  for (int kc = 0; kc < HS; kc += 32) {
    __syncthreads();
    {   // stage A: 128x32 f32 -> bf16 (16 elems / thread)
      const int flat = tid * 16;
      const int row  = flat >> 5, col = flat & 31;
      const float4* g = (const float4*)(hs + (size_t)(t0 + row) * HS + kc + col);
      cvt8(g[0], g[1], &As[row][col]);
      cvt8(g[2], g[3], &As[row][col + 8]);
    }
    {   // stage B: 64x32 f32 -> bf16 (8 elems / thread)
      const int flat = tid * 8;
      const int row  = flat >> 5, col = flat & 31;
      const float4* g = (const float4*)(W + (size_t)(o0 + row) * HS + kc + col);
      cvt8(g[0], g[1], &Bs[row][col]);
    }
    __syncthreads();

    bfx16 af0 = load_frag_a(&As[mw * 32][0],      40);
    bfx16 af1 = load_frag_a(&As[mw * 32 + 16][0], 40);
    bfx16 bf0 = load_frag_b(&Bs[nw * 32][0],      40);
    bfx16 bf1 = load_frag_b(&Bs[nw * 32 + 16][0], 40);
    acc[0][0] = wmma_bf16(af0, bf0, acc[0][0]);
    acc[0][1] = wmma_bf16(af0, bf1, acc[0][1]);
    acc[1][0] = wmma_bf16(af1, bf0, acc[1][0]);
    acc[1][1] = wmma_bf16(af1, bf1, acc[1][1]);
  }

  // epilogue: bias add, bf16 convert, scatter (single uniform branch on mat)
  if (mat == 2) {
    __bf16* vbase = Vt + (size_t)h * HD * SEQ;   // + b*NH*HD*SEQ added per elem
#pragma unroll
    for (int fr = 0; fr < 2; ++fr)
#pragma unroll
      for (int fc = 0; fc < 2; ++fc) {
        const int dcol = nw * 32 + fc * 16 + (lane & 15);
        const float bv_ = bias[o0 + dcol];
#pragma unroll
        for (int j = 0; j < 8; ++j) {
          const int m = (lane >> 4) * 8 + j;
          const int t = t0 + mw * 32 + fr * 16 + m;
          const int b = t >> 10, s = t & 1023;
          vbase[((size_t)b * NH * HD + dcol) * SEQ + s] =
              (__bf16)(acc[fr][fc][j] + bv_);
        }
      }
  } else {
    __bf16* dbase = ((mat == 0) ? Qh : Kh) + (size_t)h * SEQ * HD;
#pragma unroll
    for (int fr = 0; fr < 2; ++fr)
#pragma unroll
      for (int fc = 0; fc < 2; ++fc) {
        const int dcol = nw * 32 + fc * 16 + (lane & 15);
        const float bv_ = bias[o0 + dcol];
#pragma unroll
        for (int j = 0; j < 8; ++j) {
          const int m = (lane >> 4) * 8 + j;
          const int t = t0 + mw * 32 + fr * 16 + m;
          const int b = t >> 10, s = t & 1023;
          dbase[((size_t)b * NH * SEQ + s) * HD + dcol] =
              (__bf16)(acc[fr][fc][j] + bv_);
        }
      }
  }
}

// ---------------------------------------------------------------------------
// Kernel 2: flash-style attention with relative position bias.
// grid.x = b*16+h (64), grid.y = query block of 64 (16). 128 threads = 4 waves,
// each wave owns a 16-query strip with private online softmax.
// ---------------------------------------------------------------------------
__global__ __launch_bounds__(128) void attn_kernel(
    const __bf16* __restrict__ Qh, const __bf16* __restrict__ Kh,
    const __bf16* __restrict__ Vt,
    const float* __restrict__ dist_emb, const float* __restrict__ mask,
    float* __restrict__ out)
{
  const int bh = blockIdx.x;
  const int b  = bh >> 4, h = bh & 15;
  const int l0 = blockIdx.y * 64;

  const int tid  = threadIdx.x;
  const int wave = tid >> 5;
  const int lane = tid & 31;
  const int l0w  = l0 + wave * 16;

  __shared__ __bf16 Kt[32][72];        // key tile [r][d]; TDM pad: 32+4 dwords
  __shared__ __bf16 Vs[64][40];        // V^T tile [d][r]; TDM pad: 16+4 dwords
  __shared__ __bf16 Ew[96][72];        // dist_emb window [c][d]
  __shared__ __bf16 QEs[4][16][56];    // per-wave q*E band
  __shared__ __bf16 KEs[4][32][56];    // per-wave k*E band
  __shared__ __bf16 Ps [4][16][40];    // per-wave P tile (C->A relayout)

  const __bf16* Qb = Qh + (size_t)(b * NH + h) * SEQ * HD;
  const __bf16* Kb = Kh + (size_t)(b * NH + h) * SEQ * HD;
  const __bf16* Vb = Vt + (size_t)(b * NH + h) * HD * SEQ;

  // Q A-fragments (16 queries x 64 dim), loaded once straight from global.
  bfx16 aq0 = load_frag_a(Qb + (size_t)l0w * HD, HD);
  bfx16 aq1 = load_frag_a(Qb + (size_t)l0w * HD + 32, HD);

  fx8 acc[4];
#pragma unroll
  for (int i = 0; i < 4; ++i) acc[i] = zero8();
  fx8 mrow, lrow;
#pragma unroll
  for (int j = 0; j < 8; ++j) { mrow[j] = -1e30f; lrow[j] = 0.f; }

  for (int r0 = 0; r0 < SEQ; r0 += 32) {
    __syncthreads();   // prior tile reads (all waves) complete before re-stage

#if TDM_MODE
    if (wave == 0) {
      // K tile: 32 rows x 64 elems, row stride 64; LDS pitch 72 bf16
      tdm_load_tile_2d((unsigned)(uintptr_t)&Kt[0][0],
                       Kb + (size_t)r0 * HD,
                       /*tensor*/ 64, 32, /*tile*/ 64, 32,
                       /*stride0*/ 64, /*pad: every 32 dw*/ 4, /*+4 dw*/ 3);
      // V^T tile: 64 rows x 32 elems, row stride 1024; LDS pitch 40 bf16
      tdm_load_tile_2d((unsigned)(uintptr_t)&Vs[0][0],
                       Vb + r0,
                       /*tensor*/ 32, 64, /*tile*/ 32, 64,
                       /*stride0*/ 1024, /*pad: every 16 dw*/ 3, /*+4 dw*/ 3);
    }
#else
    for (int idx = tid; idx < 32 * 8; idx += 128) {
      const int row = idx >> 3, c8 = (idx & 7) * 8;
      *(uint4*)&Kt[row][c8] = *(const uint4*)(Kb + (size_t)(r0 + row) * HD + c8);
    }
    for (int idx = tid; idx < 64 * 4; idx += 128) {
      const int row = idx >> 2, c8 = (idx & 3) * 8;
      *(uint4*)&Vs[row][c8] = *(const uint4*)(Vb + (size_t)row * SEQ + r0 + c8);
    }
#endif

    // ---- stage dist_emb window (96x64, f32 -> bf16), all threads ----
    const int base_wg = l0 - r0 + 992;   // >= 0 by construction
    for (int idx = tid; idx < 96 * 8; idx += 128) {
      const int row = idx >> 3, c8 = (idx & 7) * 8;
      int e = base_wg + row; e = (e > 2046) ? 2046 : e;  // clamp top pad row
      const float4* g = (const float4*)(dist_emb + (size_t)e * HD + c8);
      cvt8(g[0], g[1], &Ew[row][c8]);
    }
    if (r0 + 32 < SEQ)  // pull next key tile toward L2
      __builtin_prefetch(Kb + (size_t)(r0 + 32) * HD + lane * 64, 0, 0);

#if TDM_MODE
    if (wave == 0) __builtin_amdgcn_s_wait_tensorcnt(0);
#endif
    __syncthreads();   // DMA/copies visible to all waves

    // ---- position-bias band GEMMs: QEw (16x48), KEw (32x48), K = 64 ----
#pragma unroll
    for (int ct = 0; ct < 3; ++ct) {
      bfx16 be0 = load_frag_b(&Ew[wave * 16 + ct * 16][0],  72);
      bfx16 be1 = load_frag_b(&Ew[wave * 16 + ct * 16][32], 72);
      fx8 qe = zero8();
      qe = wmma_bf16(aq0, be0, qe);
      qe = wmma_bf16(aq1, be1, qe);
      store_frag_c_bf16(&QEs[wave][0][ct * 16], 56, qe);
#pragma unroll
      for (int ar = 0; ar < 2; ++ar) {
        bfx16 ak0 = load_frag_a(&Kt[ar * 16][0],  72);
        bfx16 ak1 = load_frag_a(&Kt[ar * 16][32], 72);
        fx8 ke = zero8();
        ke = wmma_bf16(ak0, be0, ke);
        ke = wmma_bf16(ak1, be1, ke);
        store_frag_c_bf16(&KEs[wave][ar * 16][ct * 16], 56, ke);
      }
    }

    // ---- raw scores: Q (16x64) x K^T (64x32) ----
    fx8 sfr[2];
#pragma unroll
    for (int nt = 0; nt < 2; ++nt) {
      bfx16 bk0 = load_frag_b(&Kt[nt * 16][0],  72);
      bfx16 bk1 = load_frag_b(&Kt[nt * 16][32], 72);
      fx8 d0 = zero8();
      d0 = wmma_bf16(aq0, bk0, d0);
      d0 = wmma_bf16(aq1, bk1, d0);
      const int nl = nt * 16 + (lane & 15);            // key idx in tile
      const float mv = mask[(size_t)b * SEQ + r0 + nl];
#pragma unroll
      for (int j = 0; j < 8; ++j) {
        const int m = (lane >> 4) * 8 + j;
        const int c = m - nl + 31;                     // band column, 0..46
        const float qpe = (float)QEs[wave][m][c];
        const float kpe = (float)KEs[wave][nl][c];
        d0[j] = (d0[j] + qpe + kpe) * 0.125f + mv;
      }
      sfr[nt] = d0;
    }

    // ---- online softmax (rows live in 16-lane groups; wave32 shfl) ----
    fx8 tmax;
#pragma unroll
    for (int j = 0; j < 8; ++j) tmax[j] = fmaxf(sfr[0][j], sfr[1][j]);
#pragma unroll
    for (int off = 1; off < 16; off <<= 1)
#pragma unroll
      for (int j = 0; j < 8; ++j)
        tmax[j] = fmaxf(tmax[j], __shfl_xor(tmax[j], off, 32));

    fx8 mnew, scale;
#pragma unroll
    for (int j = 0; j < 8; ++j) {
      mnew[j]  = fmaxf(mrow[j], tmax[j]);
      scale[j] = __expf(mrow[j] - mnew[j]);
    }
#pragma unroll
    for (int f = 0; f < 4; ++f)
#pragma unroll
      for (int j = 0; j < 8; ++j) acc[f][j] *= scale[j];

    fx8 psum;
#pragma unroll
    for (int j = 0; j < 8; ++j) {
      sfr[0][j] = __expf(sfr[0][j] - mnew[j]);
      sfr[1][j] = __expf(sfr[1][j] - mnew[j]);
      psum[j]   = sfr[0][j] + sfr[1][j];
    }
#pragma unroll
    for (int off = 1; off < 16; off <<= 1)
#pragma unroll
      for (int j = 0; j < 8; ++j) psum[j] += __shfl_xor(psum[j], off, 32);
#pragma unroll
    for (int j = 0; j < 8; ++j) { lrow[j] = lrow[j] * scale[j] + psum[j]; mrow[j] = mnew[j]; }

    // ---- P (C layout) -> LDS -> A layout, then P @ V ----
    store_frag_c_bf16(&Ps[wave][0][0],  40, sfr[0]);
    store_frag_c_bf16(&Ps[wave][0][16], 40, sfr[1]);
    bfx16 ap = load_frag_a(&Ps[wave][0][0], 40);
#pragma unroll
    for (int dt = 0; dt < 4; ++dt) {
      bfx16 bv = load_frag_b(&Vs[dt * 16][0], 40);
      acc[dt] = wmma_bf16(ap, bv, acc[dt]);
    }
  }

  // ---- normalize and write context: out[b, s, h*64 + d] ----
  fx8 inv;
#pragma unroll
  for (int j = 0; j < 8; ++j) inv[j] = 1.0f / lrow[j];
#pragma unroll
  for (int dt = 0; dt < 4; ++dt) {
    const int d = dt * 16 + (lane & 15);
#pragma unroll
    for (int j = 0; j < 8; ++j) {
      const int m = (lane >> 4) * 8 + j;
      const int s = l0w + m;
      out[((size_t)(b * SEQ + s)) * HS + h * HD + d] = acc[dt][j] * inv[j];
    }
  }
}

// ---------------------------------------------------------------------------
extern "C" void kernel_launch(void* const* d_in, const int* in_sizes, int n_in,
                              void* d_out, int out_size, void* d_ws, size_t ws_size,
                              hipStream_t stream) {
  const float* hs   = (const float*)d_in[0];
  const float* mask = (const float*)d_in[1];
  const float* Wq   = (const float*)d_in[2];
  const float* bq   = (const float*)d_in[3];
  const float* Wk   = (const float*)d_in[4];
  const float* bk   = (const float*)d_in[5];
  const float* Wv   = (const float*)d_in[6];
  const float* bv   = (const float*)d_in[7];
  const float* dist = (const float*)d_in[8];
  float* out = (float*)d_out;

  const size_t per = (size_t)4 * NH * SEQ * HD;   // 4 Mi elems each
  __bf16* Qh = (__bf16*)d_ws;
  __bf16* Kh = Qh + per;
  __bf16* Vt = Kh + per;

  // Kernel 1: M=4096 tokens / 128, N = 3 matrices x 16 heads (64 cols each)
  qkv_proj_kernel<<<dim3(32, 48), 256, 0, stream>>>(
      hs, Wq, Wk, Wv, bq, bk, bv, Qh, Kh, Vt);

  // Kernel 2: 64 (b,h) pairs x 16 query blocks of 64
  attn_kernel<<<dim3(64, 16), 128, 0, stream>>>(Qh, Kh, Vt, dist, mask, out);
}
